// MockUPIQAL_6158983102922
// MI455X (gfx1250) — matrix operand order repaired and checked by reading, not compiled, not run on hardware
//
#include <hip/hip_runtime.h>

// MI455X (gfx1250, wave32). Memory-bound pipeline (~500MB traffic @ 23.3TB/s
// ~= 20us). Fused into: init + elementwise/reduce + score + tiled-stencil.
// 7x7 box filter runs on the WMMA units as separable band-matrix GEMMs
// (v_wmma_f32_16x16x32_f16). Interior halo tiles are fetched with CDNA5
// GLOBAL_LOAD_ASYNC_TO_LDS (ASYNCcnt) instead of a global->VGPR->ds_store
// round trip; border tiles fall back to the zero-padding VGPR path.

typedef __attribute__((ext_vector_type(16))) _Float16 v16h;
typedef __attribute__((ext_vector_type(8)))  float    v8f;
typedef __attribute__((ext_vector_type(4)))  float    v4f;

constexpr int  BB = 4;
constexpr int  H  = 1080;
constexpr int  W  = 1920;
constexpr long HW = (long)H * (long)W;   // 2,073,600

// ---------------------------------------------------------------- k_init
__global__ void k_init(unsigned int* red) {
    if (threadIdx.x < 32) red[threadIdx.x] = 0u;   // 128B: maxes + u64 sums
}

// ---------------------------------------------------------------- k_prep
// Per thread: 4 consecutive pixels (v4f). Writes unnormalized diff->ch0,
// colordiff->ch1, grays->ws. Reductions: bitwise atomicMax (floats >= 0 so
// IEEE bits are monotonic) and u64 fixed-point atomicAdd (deterministic).
__global__ __launch_bounds__(256) void k_prep(
    const float* __restrict__ ref, const float* __restrict__ tgt,
    float* __restrict__ diag, float* __restrict__ refg, float* __restrict__ tgtg,
    unsigned int* __restrict__ red, unsigned long long* __restrict__ sums)
{
    const int b  = blockIdx.y;
    const long p = ((long)blockIdx.x * 256 + threadIdx.x) * 4;
    const float* rb = ref + (long)b * 3 * HW + p;
    const float* tb = tgt + (long)b * 3 * HW + p;

    v4f r0 = *(const v4f*)(rb);
    v4f r1 = *(const v4f*)(rb + HW);
    v4f r2 = *(const v4f*)(rb + 2 * HW);
    v4f t0 = *(const v4f*)(tb);
    v4f t1 = *(const v4f*)(tb + HW);
    v4f t2 = *(const v4f*)(tb + 2 * HW);

    v4f diffv, cdfv, rgv, tgv;
    float lmaxd = 0.f, lmaxc = 0.f, lsum = 0.f;
#pragma unroll
    for (int j = 0; j < 4; j++) {
        float d0 = r0[j] - t0[j], d1 = r1[j] - t1[j], d2 = r2[j] - t2[j];
        float df = sqrtf(d0 * d0 + d1 * d1 + d2 * d2);
        float cf = (fabsf(d0) + fabsf(d1) + fabsf(d2)) * (1.f / 3.f);
        diffv[j] = df;
        cdfv[j]  = cf;
        rgv[j] = 0.299f * r0[j] + 0.587f * r1[j] + 0.114f * r2[j];
        tgv[j] = 0.299f * t0[j] + 0.587f * t1[j] + 0.114f * t2[j];
        lmaxd = fmaxf(lmaxd, df);
        lmaxc = fmaxf(lmaxc, cf);
        lsum += df;
    }
    const long ob = (long)b * 5 * HW + p;
    *(v4f*)(diag + ob)            = diffv;   // ch0 (normalized later)
    *(v4f*)(diag + ob + HW)       = cdfv;    // ch1 (normalized later)
    *(v4f*)(refg + (long)b * HW + p) = rgv;
    *(v4f*)(tgtg + (long)b * HW + p) = tgv;

    // wave32 tree reduce (deterministic), one atomic per wave
#pragma unroll
    for (int off = 16; off > 0; off >>= 1) {
        lmaxd = fmaxf(lmaxd, __shfl_down(lmaxd, off, 32));
        lmaxc = fmaxf(lmaxc, __shfl_down(lmaxc, off, 32));
        lsum += __shfl_down(lsum, off, 32);
    }
    if ((threadIdx.x & 31) == 0) {
        atomicMax(&red[0], __float_as_uint(lmaxd));
        atomicMax(&red[1], __float_as_uint(lmaxc));
        atomicAdd(&sums[b], (unsigned long long)(lsum * 1048576.f + 0.5f));
    }
}

// ---------------------------------------------------------------- k_score
__global__ void k_score(const unsigned int* __restrict__ red,
                        const unsigned long long* __restrict__ sums,
                        float* __restrict__ score)
{
    int b = threadIdx.x;
    if (b < BB) {
        float maxd = __uint_as_float(red[0]);
        float mean = (float)((double)sums[b] * (1.0 / 1048576.0) / (double)HW);
        float an   = mean / (maxd + 1e-12f);
        score[b]   = fminf(fmaxf(1.f - an, 0.f), 1.f);
    }
}

// ---------------------------------------------------------------- k_tile
// 16x16 output tile. 22x22 gray halo tiles in LDS (halo 3 covers 7x7 box,
// and Sobel on the +/-2 extended region needed by the 5x5 dilation).
// Box filter via WMMA: Hs = G(22x22) * Band(22x16); mu = Band(16x22) * Hs.
constexpr int TS = 16, HLO = 3, EX = 20, GW = 22, GSTR = 24;

__global__ __launch_bounds__(256) void k_tile(
    const float* __restrict__ refg, const float* __restrict__ tgtg,
    const unsigned int* __restrict__ red, float* __restrict__ diag)
{
    __shared__ float    sR[GW][GSTR];
    __shared__ float    sT[GW][GSTR];
    __shared__ float    sE[EX][EX + 1];
    __shared__ _Float16 sHs[2][GW][TS];
    __shared__ float    sMu[2][TS][TS];

    const int tid  = threadIdx.x;
    const int lane = tid & 31;
    const int wave = tid >> 5;
    const int b    = blockIdx.z;
    const int x0   = blockIdx.x * TS;
    const int y0   = blockIdx.y * TS;
    const float* rg = refg + (long)b * HW;
    const float* tg = tgtg + (long)b * HW;

    // ---- load 22x22 halo tiles.
    // Interior tiles: CDNA5 async DMA global->LDS (no VGPR staging).
    // Border tiles: VGPR path with zero padding (conv zero-pad semantics).
    const bool interior = (x0 >= HLO) && (y0 >= HLO) &&
                          (x0 + TS + HLO <= W) && (y0 + TS + HLO <= H);
    if (interior) {
        for (int i = tid; i < GW * GW; i += 256) {
            int ly = i / GW, lx = i - ly * GW;
            long o = (long)(y0 + ly - HLO) * W + (x0 + lx - HLO);
            unsigned int ldsR = (unsigned int)(uintptr_t)&sR[ly][lx];
            unsigned int ldsT = (unsigned int)(uintptr_t)&sT[ly][lx];
            unsigned long long gaR = (unsigned long long)(uintptr_t)(rg + o);
            unsigned long long gaT = (unsigned long long)(uintptr_t)(tg + o);
            asm volatile("global_load_async_to_lds_b32 %0, %1, off"
                         :: "v"(ldsR), "v"(gaR) : "memory");
            asm volatile("global_load_async_to_lds_b32 %0, %1, off"
                         :: "v"(ldsT), "v"(gaT) : "memory");
        }
        asm volatile("s_wait_asynccnt 0x0" ::: "memory");
    } else {
        for (int i = tid; i < GW * GW; i += 256) {
            int ly = i / GW, lx = i - ly * GW;
            int gx = x0 + lx - HLO, gy = y0 + ly - HLO;
            bool in = (gx >= 0) & (gx < W) & (gy >= 0) & (gy < H);
            long o  = (long)gy * W + gx;
            sR[ly][lx] = in ? rg[o] : 0.f;
            sT[ly][lx] = in ? tg[o] : 0.f;
        }
    }
    __syncthreads();

    // ---- Sobel edges on extended region (center-2 .. center+17);
    //      out-of-image -> 0 (dilation pad is -inf i.e. non-contributing)
    for (int i = tid; i < EX * EX; i += 256) {
        int ey = i / EX, ex = i - ey * EX;
        int hy = ey + 1, hx = ex + 1;
        float a00 = sT[hy-1][hx-1], a01 = sT[hy-1][hx], a02 = sT[hy-1][hx+1];
        float a10 = sT[hy  ][hx-1],                     a12 = sT[hy  ][hx+1];
        float a20 = sT[hy+1][hx-1], a21 = sT[hy+1][hx], a22 = sT[hy+1][hx+1];
        float gxv = fabsf((a02 + 2.f*a12 + a22) - (a00 + 2.f*a10 + a20));
        float gyv = fabsf((a20 + 2.f*a21 + a22) - (a00 + 2.f*a01 + a02));
        float gm  = sqrtf(gxv * gxv + gyv * gyv);
        int gxp = x0 + ex - 2, gyp = y0 + ey - 2;
        bool in = (gxp >= 0) & (gxp < W) & (gyp >= 0) & (gyp < H);
        sE[ey][ex] = (in && gm > 0.15f) ? 1.f : 0.f;
    }

    // ---- WMMA box filter. All 8 waves run (EXEC all-ones at each WMMA,
    //      per ISA restriction); even waves -> ref, odd waves -> tgt,
    //      waves 0/1 commit results to LDS.
    const int img  = wave & 1;
    const float (*sG)[GSTR] = img ? sT : sR;
    const int colN   = lane & 15;              // N (and A-row) owned by lane
    const int kbaseB = (lane < 16) ? 0 : 16;   // unused; B takes the same K as A

    v16h a1, a2, bm;
#pragma unroll
    for (int e = 0; e < 16; e++) {
        // A-layout K for 16-bit A 16x32 (ISA 7.12.2)
        int K  = (e < 8 ? e : e + 8) + ((lane < 16) ? 0 : 8);
        int rA = colN;            // rows 0..15
        int rB = 16 + colN;       // rows 16..31 (valid < 22)
        a1[e] = (K < GW) ? (_Float16)sG[rA][K] : (_Float16)0.f;
        a2[e] = (rB < GW && K < GW) ? (_Float16)sG[rB][K] : (_Float16)0.f;
        int Kb = K;               // B-layout K (same order as A)
        bm[e] = (_Float16)((Kb >= colN && Kb <= colN + 6) ? 1.f : 0.f);
    }
    v8f z = {};
    v8f d1 = __builtin_amdgcn_wmma_f32_16x16x32_f16(false, a1, false, bm,
                                                    (short)0, z, false, false);
    v8f d2 = __builtin_amdgcn_wmma_f32_16x16x32_f16(false, a2, false, bm,
                                                    (short)0, z, false, false);
    if (wave < 2) {               // commit Hs(22x16): D layout -> LDS
        if (lane < 16) {
#pragma unroll
            for (int k = 0; k < 8; k++) sHs[img][k][colN]      = (_Float16)d1[k];
#pragma unroll
            for (int k = 0; k < 6; k++) sHs[img][16 + k][colN] = (_Float16)d2[k];
        } else {
#pragma unroll
            for (int k = 0; k < 8; k++) sHs[img][8 + k][colN]  = (_Float16)d1[k];
        }
    }
    __syncthreads();

    // ---- pass 2: mu = Band(16x22) * Hs(22x16)
    v16h ab, bh;
#pragma unroll
    for (int e = 0; e < 16; e++) {
        int K  = (e < 8 ? e : e + 8) + ((lane < 16) ? 0 : 8);
        ab[e]  = (_Float16)((K >= colN && K <= colN + 6) ? 1.f : 0.f);
        int Kb = K;
        bh[e]  = (Kb < GW) ? sHs[img][Kb][colN] : (_Float16)0.f;
    }
    v8f dmu = __builtin_amdgcn_wmma_f32_16x16x32_f16(false, ab, false, bh,
                                                     (short)0, z, false, false);
    if (wave < 2) {
        const float inv49 = 1.f / 49.f;
        if (lane < 16) {
#pragma unroll
            for (int k = 0; k < 8; k++) sMu[img][k][colN]     = dmu[k] * inv49;
        } else {
#pragma unroll
            for (int k = 0; k < 8; k++) sMu[img][k + 8][colN] = dmu[k] * inv49;
        }
    }
    __syncthreads();

    // ---- per-pixel assembly
    const int tx = tid & 15, ty = tid >> 4;
    const int gx = x0 + tx, gy = y0 + ty;
    if (gx < W && gy < H) {
        const float invmaxd = 1.f / (__uint_as_float(red[0]) + 1e-12f);
        const float invmaxc = 1.f / (__uint_as_float(red[1]) + 1e-12f);
        const long base = (long)b * 5 * HW + (long)gy * W + gx;

        float diffv = diag[base];
        float an    = diffv * invmaxd;
        float cdf   = diag[base + HW];

        float mur = sMu[0][ty][tx], mut = sMu[1][ty][tx];
        float ssim = (2.f * mur * mut + 1e-4f) / (mur * mur + mut * mut + 1e-4f);

        // Sobel at the center pixel (for blocking mask)
        int hy = ty + HLO, hx = tx + HLO;
        float a00 = sT[hy-1][hx-1], a01 = sT[hy-1][hx], a02 = sT[hy-1][hx+1];
        float a10 = sT[hy  ][hx-1],                     a12 = sT[hy  ][hx+1];
        float a20 = sT[hy+1][hx-1], a21 = sT[hy+1][hx], a22 = sT[hy+1][hx+1];
        float gxv = fabsf((a02 + 2.f*a12 + a22) - (a00 + 2.f*a10 + a20));
        float gyv = fabsf((a20 + 2.f*a21 + a22) - (a00 + 2.f*a01 + a02));

        float blk = ((gx & 7) == 0) ? gxv : 0.f;
        if ((gy & 7) == 0) blk = fmaxf(blk, gyv);
        float blocking = (blk > 0.05f) ? 1.f : 0.f;

        float dil = 0.f;
#pragma unroll
        for (int dy = 0; dy < 5; dy++)
#pragma unroll
            for (int dx = 0; dx < 5; dx++)
                dil = fmaxf(dil, sE[ty + dy][tx + dx]);
        float edge    = sE[ty + 2][tx + 2];
        float ringing = fmaxf(dil - edge, 0.f) * an;

        diag[base]          = an;               // anomaly
        diag[base + HW]     = cdf * invmaxc;    // color_map
        diag[base + 2 * HW] = ssim;             // ssim_approx
        diag[base + 3 * HW] = blocking;         // blocking
        diag[base + 4 * HW] = ringing;          // ringing
    }
}

// ---------------------------------------------------------------- launch
extern "C" void kernel_launch(void* const* d_in, const int* in_sizes, int n_in,
                              void* d_out, int out_size, void* d_ws, size_t ws_size,
                              hipStream_t stream)
{
    const float* ref = (const float*)d_in[0];
    const float* tgt = (const float*)d_in[1];
    float* out  = (float*)d_out;
    float* diag = out + BB;                      // score(4) then (B,5,H,W)

    unsigned int*       red  = (unsigned int*)d_ws;                 // [0..31]
    unsigned long long* sums = (unsigned long long*)((char*)d_ws + 64);
    float* refg = (float*)((char*)d_ws + 128);
    float* tgtg = refg + (size_t)BB * HW;

    k_init<<<1, 32, 0, stream>>>(red);
    k_prep<<<dim3((unsigned)(HW / 1024), BB), 256, 0, stream>>>(
        ref, tgt, diag, refg, tgtg, red, sums);
    k_score<<<1, 32, 0, stream>>>(red, sums, out);
    k_tile<<<dim3(W / TS, (H + TS - 1) / TS, BB), 256, 0, stream>>>(
        refg, tgtg, red, diag);
}
